// MultiHeadAttention_31920196944521
// MI455X (gfx1250) — matrix operand order repaired and checked by reading, not compiled, not run on hardware
//
#include <hip/hip_runtime.h>
#include <hip/hip_bf16.h>

// MHA forward for B=2,T=2048,E=1024,H=16,D=64 on gfx1250 (wave32, WMMA).
typedef __attribute__((ext_vector_type(16))) _Float16 v16h;
typedef __attribute__((ext_vector_type(8)))  _Float16 v8h;
typedef __attribute__((ext_vector_type(8)))  float    v8f;

constexpr int BB = 2, TT = 2048, EE = 1024, HH = 16, DD = 64;

__device__ __forceinline__ v8f wmma_f16(v16h a, v16h b, v8f c) {
  // v_wmma_f32_16x16x32_f16  (neg_a, A, neg_b, B, c_mod, C, reuse_a, reuse_b)
  return __builtin_amdgcn_wmma_f32_16x16x32_f16(false, a, false, b, (short)0, c,
                                                false, false);
}

// Build A-fragment (16x32 f16) for one K-step. `row` points at A[m][k0]
// (m = lane%16), kb = 8*(lane/16). Elements 0..15 = K 2*kb..2*kb+15
__device__ __forceinline__ v16h make_afrag(const _Float16* row, int kb) {
  v8h lo = *(const v8h*)(row + 2 * kb);
  v8h hi = *(const v8h*)(row + 2 * kb + 8);
  return __builtin_shufflevector(lo, hi, 0, 1, 2, 3, 4, 5, 6, 7,
                                 8, 9, 10, 11, 12, 13, 14, 15);
}

__global__ void mha_cvt_f16_kernel(const float* __restrict__ in,
                                   _Float16* __restrict__ out, int n) {
  int i = blockIdx.x * blockDim.x + threadIdx.x;
  if (i < n) out[i] = (_Float16)in[i];
}

// in: [nmat, M, N] fp32 -> out: [nmat, N, M] f16 (transpose inner 2 dims)
__global__ void mha_transpose_cvt_kernel(const float* __restrict__ in,
                                         _Float16* __restrict__ out,
                                         int M, int N, int n) {
  int i = blockIdx.x * blockDim.x + threadIdx.x;
  if (i >= n) return;
  int mn = M * N;
  int mat = i / mn, r = (i % mn) / N, c = i % N;
  out[mat * mn + c * M + r] = (_Float16)in[i];
}

// C[t0:t0+16, n0:n0+16] = xh[b] @ W^T-row-major, per (b,h). 4 waves cover D=64.
// TRANSPOSE_OUT=1 writes out[b,h,d,t] (for V), else out[b,h,t,d].
template <int TRANSPOSE_OUT>
__global__ void mha_qkv_gemm_kernel(const _Float16* __restrict__ xh,
                                    const _Float16* __restrict__ wT,
                                    _Float16* __restrict__ out) {
  int lane = threadIdx.x & 31, wave = threadIdx.x >> 5;
  int tTile = blockIdx.x % (TT / 16);
  int bh = blockIdx.x / (TT / 16);
  int h = bh % HH, b = bh / HH;
  int t0 = tTile * 16, n0 = wave * 16;
  int lo = lane & 15, hi16 = lane >> 4, kb = hi16 << 3;

  const _Float16* arow = xh + (size_t)(b * TT + t0 + lo) * EE;
  const _Float16* brow = wT + (size_t)(h * DD + n0 + lo) * EE + (hi16 << 4);

  v8f c = {};
#pragma unroll 4
  for (int k0 = 0; k0 < EE; k0 += 32) {
    v16h a = make_afrag(arow + k0, kb);
    v16h bf = *(const v16h*)(brow + k0);
    c = wmma_f16(a, bf, c);
  }
  int mb = hi16 << 3;
#pragma unroll
  for (int r = 0; r < 8; ++r) {
    int t = t0 + mb + r, d = n0 + lo;
    _Float16 v = (_Float16)c[r];
    if (TRANSPOSE_OUT)
      out[(size_t)((b * HH + h) * DD + d) * TT + t] = v;
    else
      out[(size_t)((b * HH + h) * TT + t) * DD + d] = v;
  }
}

// Flash attention: one wave per 16 q-rows; s-blocks of 32; online softmax.
// qh,kh: [B,H,T,D] f16; vth: [B,H,D,T] f16; attnh: [B,T,E] f16 (heads concat).
// waves_per_eu target 1: let the allocator use enough VGPRs to avoid spills.
__global__ void __launch_bounds__(128, 1)
mha_attn_kernel(const _Float16* __restrict__ qh,
                const _Float16* __restrict__ kh,
                const _Float16* __restrict__ vth,
                _Float16* __restrict__ attnh) {
  __shared__ __align__(16) _Float16 pbuf[4][16 * 32];
  int lane = threadIdx.x & 31, wave = threadIdx.x >> 5;
  int rTile = blockIdx.x % (TT / 64);
  int bh = blockIdx.x / (TT / 64);
  int h = bh % HH, b = bh / HH;
  int t0 = rTile * 64 + wave * 16;
  int lo = lane & 15, hi16 = lane >> 4, kb = hi16 << 3;

  const _Float16* qrow = qh + (size_t)((b * HH + h) * TT + t0 + lo) * DD;
  const _Float16* kbase = kh + (size_t)(b * HH + h) * TT * DD;
  const _Float16* vbase = vth + (size_t)(b * HH + h) * DD * TT;

  v16h aq0 = make_afrag(qrow, kb);        // K (=d) 0..31
  v16h aq1 = make_afrag(qrow + 32, kb);   // K (=d) 32..63

  float mrow[8], lrow[8];
  v8f o[4];
  v8f zero = {};
#pragma unroll
  for (int r = 0; r < 8; ++r) { mrow[r] = -INFINITY; lrow[r] = 0.0f; }
#pragma unroll
  for (int j = 0; j < 4; ++j) o[j] = zero;

  const float scale = 0.03125f;  // E^-0.5 = 1/32 (ref scales by n_embd^-0.5)
  int s_end = t0 + 16;
  int mb = hi16 << 3;
  int tbase = t0 + mb;  // row index of this lane's VGPR r is tbase + r

  for (int s0 = 0; s0 < s_end; s0 += 32) {
    float p[2][8];
#pragma unroll
    for (int nsub = 0; nsub < 2; ++nsub) {
      int srow = s0 + nsub * 16 + lo;
      const _Float16* kr = kbase + (size_t)srow * DD + (hi16 << 4);
      v16h bk0 = *(const v16h*)kr;          // d 0..31 slice
      v16h bk1 = *(const v16h*)(kr + 32);   // d 32..63 slice
      v8f sc = zero;
      sc = wmma_f16(aq0, bk0, sc);
      sc = wmma_f16(aq1, bk1, sc);
      // causal: keep iff srow <= tbase + r  <=>  (srow - tbase) <= r
      // comparing against immediate r keeps thresholds out of VGPRs
      int dlim = srow - tbase;
#pragma unroll
      for (int r = 0; r < 8; ++r)
        p[nsub][r] = (dlim <= r) ? sc[r] * scale : -INFINITY;
    }
    // online softmax: per-row max/sum (row m lives on 16 lanes of a half-wave)
    float mnew[8], alpha[8];
#pragma unroll
    for (int r = 0; r < 8; ++r) {
      float v = fmaxf(p[0][r], p[1][r]);
      v = fmaxf(v, __shfl_xor(v, 8, 32));
      v = fmaxf(v, __shfl_xor(v, 4, 32));
      v = fmaxf(v, __shfl_xor(v, 2, 32));
      v = fmaxf(v, __shfl_xor(v, 1, 32));
      mnew[r] = fmaxf(mrow[r], v);
    }
#pragma unroll
    for (int r = 0; r < 8; ++r) {
      p[0][r] = __expf(p[0][r] - mnew[r]);
      p[1][r] = __expf(p[1][r] - mnew[r]);
      float s = p[0][r] + p[1][r];
      s += __shfl_xor(s, 8, 32);
      s += __shfl_xor(s, 4, 32);
      s += __shfl_xor(s, 2, 32);
      s += __shfl_xor(s, 1, 32);
      alpha[r] = __expf(mrow[r] - mnew[r]);
      lrow[r] = lrow[r] * alpha[r] + s;
      mrow[r] = mnew[r];
    }
#pragma unroll
    for (int j = 0; j < 4; ++j)
#pragma unroll
      for (int r = 0; r < 8; ++r) o[j][r] *= alpha[r];

    // C-fragment -> A-fragment reshape for P via LDS (wave-private 1KB)
#pragma unroll
    for (int j = 0; j < 2; ++j)
#pragma unroll
      for (int r = 0; r < 8; ++r)
        pbuf[wave][(mb + r) * 32 + j * 16 + lo] = (_Float16)p[j][r];
    asm volatile("s_wait_dscnt 0" ::: "memory");
    v16h ap = make_afrag(&pbuf[wave][lo * 32], kb);

    // O += P @ V  (B-fragment contiguous from V^T [D,T])
#pragma unroll
    for (int j = 0; j < 4; ++j) {
      const _Float16* vr = vbase + (size_t)(j * 16 + lo) * TT + s0 + (hi16 << 4);
      v16h bv = *(const v16h*)vr;
      o[j] = wmma_f16(ap, bv, o[j]);
    }
  }

#pragma unroll
  for (int r = 0; r < 8; ++r) lrow[r] = 1.0f / lrow[r];
#pragma unroll
  for (int j = 0; j < 4; ++j)
#pragma unroll
    for (int r = 0; r < 8; ++r) {
      int t = tbase + r;
      int col = h * DD + j * 16 + lo;
      attnh[(size_t)(b * TT + t) * EE + col] = (_Float16)(o[j][r] * lrow[r]);
    }
}

// Y[m, n] = attnh[m,:] @ Wproj[:, n] + bproj[n]   (fp32 out)
__global__ void mha_proj_kernel(const _Float16* __restrict__ ah,
                                const _Float16* __restrict__ wpT,
                                const float* __restrict__ bias,
                                float* __restrict__ y) {
  int lane = threadIdx.x & 31, wave = threadIdx.x >> 5;
  int nTile = blockIdx.x % (EE / 64);
  int m0 = (blockIdx.x / (EE / 64)) * 16;
  int n0 = nTile * 64 + wave * 16;
  int lo = lane & 15, hi16 = lane >> 4, kb = hi16 << 3;

  const _Float16* arow = ah + (size_t)(m0 + lo) * EE;
  const _Float16* brow = wpT + (size_t)(n0 + lo) * EE + (hi16 << 4);

  v8f c = {};
#pragma unroll 4
  for (int k0 = 0; k0 < EE; k0 += 32) {
    v16h a = make_afrag(arow + k0, kb);
    v16h bf = *(const v16h*)(brow + k0);
    c = wmma_f16(a, bf, c);
  }
  float bn = bias[n0 + lo];
  int mb = hi16 << 3;
#pragma unroll
  for (int r = 0; r < 8; ++r)
    y[(size_t)(m0 + mb + r) * EE + n0 + lo] = c[r] + bn;
}

extern "C" void kernel_launch(void* const* d_in, const int* in_sizes, int n_in,
                              void* d_out, int out_size, void* d_ws,
                              size_t ws_size, hipStream_t stream) {
  (void)in_sizes; (void)n_in; (void)out_size; (void)ws_size;
  const float* x     = (const float*)d_in[0];  // [B,T,E]
  const float* Wq    = (const float*)d_in[1];  // [H,E,D]
  const float* Wk    = (const float*)d_in[2];
  const float* Wv    = (const float*)d_in[3];
  const float* Wp    = (const float*)d_in[4];  // [E,E]
  const float* bproj = (const float*)d_in[5];  // [E]

  size_t off = 0;
  auto take = [&](size_t bytes) -> char* {
    char* p = (char*)d_ws + off;
    off += (bytes + 255) & ~(size_t)255;
    return p;
  };
  _Float16* xh    = (_Float16*)take((size_t)BB * TT * EE * 2);
  _Float16* wqT   = (_Float16*)take((size_t)HH * DD * EE * 2);
  _Float16* wkT   = (_Float16*)take((size_t)HH * DD * EE * 2);
  _Float16* wvT   = (_Float16*)take((size_t)HH * DD * EE * 2);
  _Float16* wpT   = (_Float16*)take((size_t)EE * EE * 2);
  _Float16* qh    = (_Float16*)take((size_t)BB * HH * TT * DD * 2);
  _Float16* kh    = (_Float16*)take((size_t)BB * HH * TT * DD * 2);
  _Float16* vth   = (_Float16*)take((size_t)BB * HH * TT * DD * 2);
  _Float16* attnh = (_Float16*)take((size_t)BB * TT * EE * 2);

  int nx = BB * TT * EE;
  mha_cvt_f16_kernel<<<nx / 256, 256, 0, stream>>>(x, xh, nx);
  int nw = HH * EE * DD;
  mha_transpose_cvt_kernel<<<nw / 256, 256, 0, stream>>>(Wq, wqT, EE, DD, nw);
  mha_transpose_cvt_kernel<<<nw / 256, 256, 0, stream>>>(Wk, wkT, EE, DD, nw);
  mha_transpose_cvt_kernel<<<nw / 256, 256, 0, stream>>>(Wv, wvT, EE, DD, nw);
  int np = EE * EE;
  mha_transpose_cvt_kernel<<<np / 256, 256, 0, stream>>>(Wp, wpT, EE, EE, np);

  dim3 gqkv(BB * HH * (TT / 16));  // 4096 blocks, 4 waves each (16x64 tile)
  mha_qkv_gemm_kernel<0><<<gqkv, 128, 0, stream>>>(xh, wqT, qh);
  mha_qkv_gemm_kernel<0><<<gqkv, 128, 0, stream>>>(xh, wkT, kh);
  mha_qkv_gemm_kernel<1><<<gqkv, 128, 0, stream>>>(xh, wvT, vth);

  mha_attn_kernel<<<BB * HH * (TT / 64), 128, 0, stream>>>(qh, kh, vth, attnh);

  mha_proj_kernel<<<(BB * TT / 16) * (EE / 64), 128, 0, stream>>>(
      attnh, wpT, bproj, (float*)d_out);
}
